// Attention_3762391351934
// MI455X (gfx1250) — matrix operand order repaired and checked by reading, not compiled, not run on hardware
//
#include <hip/hip_runtime.h>
#include <hip/hip_bf16.h>

typedef __attribute__((ext_vector_type(16))) _Float16 v16h;
typedef __attribute__((ext_vector_type(8)))  float    v8f;

#define B_   2
#define N_   2048
#define C_   1024
#define H_   16
#define DH_  64

// ---------------------------------------------------------------------------
// Generic WMMA GEMM:  C[M,Nc] = A[M,K] * W[Nc,K]^T (+ bias)
// block = 256 threads (8 waves). Block tile 64(M) x 128(N), K-step 32.
// Each wave owns a 32x32 subtile = 2x2 WMMA 16x16x32 (f16 in, f32 acc).
// LDS tiles are stored in *fragment-linear* order: each lane's 16-half
// fragment is contiguous (32B), so fragment loads are 2x ds_load_b128 with
// no VALU repacking. The layout inversion is done once on the write side.
// M % 64 == 0, Nc % 128 == 0, K % 32 == 0 (true for all uses here).
// ---------------------------------------------------------------------------
__global__ __launch_bounds__(256) void gemm_wmma_kernel(
    const float* __restrict__ A, const float* __restrict__ W,
    const float* __restrict__ bias, float* __restrict__ C,
    int M, int Nc, int K)
{
    // As[sub][lane][j]: sub = m/16 (4 subtiles of the 64-row slab),
    //   lane = 16*half + (m%16), j = A-fragment element index (K per ISA map)
    // Bs[sub][lane][j]: sub = n/16 (8 subtiles of the 128-col slab),
    //   lane = 16*half + (n%16), j: k = (j&7) + 8*half + 16*(j>>3)
    __shared__ __align__(32) _Float16 As[4][32][16];   // 4 KB
    __shared__ __align__(32) _Float16 Bs[8][32][16];   // 8 KB

    const int tid  = threadIdx.x;
    const int lane = tid & 31;
    const int wv   = tid >> 5;           // 0..7
    const int wm   = (wv >> 2) * 2;      // A-subtile base: 0 or 2
    const int wn   = (wv & 3) * 2;       // B-subtile base: 0,2,4,6
    const int half = lane >> 4;          // 0/1
    const int l15  = lane & 15;

    const int m0 = blockIdx.y * 64;
    const int n0 = blockIdx.x * 128;

    v8f acc[2][2];
    {
        v8f z = {};
        acc[0][0] = z; acc[0][1] = z; acc[1][0] = z; acc[1][1] = z;
    }

    for (int k0 = 0; k0 < K; k0 += 32) {
        // --- stage A slab (64 rows x 32 k), fp32 -> f16, coalesced on k ---
        #pragma unroll
        for (int i = tid; i < 64 * 32; i += 256) {
            int row = i >> 5, kk = i & 31;
            float v = A[(size_t)(m0 + row) * K + k0 + kk];
            // invert K = base(j>>1) + 8*h + (j&1)
            int h   = (kk >> 3) & 1;
            int rem = kk - 8 * h;                    // {0..7} u {16..23}
            int jv  = (rem < 8) ? rem : (rem - 8);   // 2*i_v + e
            As[row >> 4][h * 16 + (row & 15)][jv] = (_Float16)v;
        }
        // --- stage W slab (128 rows x 32 k): B[k][n] = W[n0+n][k0+k] ---
        #pragma unroll
        for (int i = tid; i < 128 * 32; i += 256) {
            int nn = i >> 5, kk = i & 31;
            float v = W[(size_t)(n0 + nn) * K + k0 + kk];
            // table order: h = (kk>>3)&1, j = (kk&7) + 8*(kk>>4)
            Bs[nn >> 4][((kk >> 3) & 1) * 16 + (nn & 15)][(kk & 7) | ((kk >> 4) << 3)] = (_Float16)v;
        }
        __syncthreads();

        // --- contiguous 32B fragment loads (2x ds_load_b128 each) ---
        v16h af[2], bf[2];
        #pragma unroll
        for (int s = 0; s < 2; ++s) {
            af[s] = *(const v16h*)&As[wm + s][lane][0];
            bf[s] = *(const v16h*)&Bs[wn + s][lane][0];
        }

        #pragma unroll
        for (int sm = 0; sm < 2; ++sm)
            #pragma unroll
            for (int sn = 0; sn < 2; ++sn)
                acc[sm][sn] = __builtin_amdgcn_wmma_f32_16x16x32_f16(
                    false, af[sm], false, bf[sn],
                    (short)0, acc[sm][sn], false, false);

        __syncthreads();
    }

    // --- epilogue: D layout: vgpr r -> m = r + 8*half ; n = lane%16 ---
    #pragma unroll
    for (int sm = 0; sm < 2; ++sm) {
        #pragma unroll
        for (int sn = 0; sn < 2; ++sn) {
            #pragma unroll
            for (int r = 0; r < 8; ++r) {
                int m = m0 + (wm + sm) * 16 + r + 8 * half;
                int n = n0 + (wn + sn) * 16 + l15;
                float v = acc[sm][sn][r];
                if (bias) v += bias[n];
                C[(size_t)m * Nc + n] = v;
            }
        }
    }
}

// ---------------------------------------------------------------------------
// sumV[b,h,e] = sum_n V[b,h,n,e]   (V lives inside qkv buffer at +2C)
// ---------------------------------------------------------------------------
__global__ void sumv_kernel(const float* __restrict__ qkv, float* __restrict__ sumV)
{
    int idx = blockIdx.x * blockDim.x + threadIdx.x;     // B*H*DH = 2048
    if (idx >= B_ * H_ * DH_) return;
    int e = idx & (DH_ - 1);
    int h = (idx >> 6) & (H_ - 1);
    int b = idx >> 10;
    const float* base = qkv + (size_t)b * N_ * 3 * C_ + 2 * C_ + h * DH_ + e;
    float s = 0.f;
    for (int n = 0; n < N_; ++n) s += base[(size_t)n * 3 * C_];
    sumV[idx] = s;
}

// ---------------------------------------------------------------------------
// Band attention with rank-1 out-of-band correction.
// One wave32 per (b,h,n). softmax over N entries where out-of-band logits
// are exactly 1e-9 -> constant prob pnb; out = sum_band p*v + pnb*(sumV - bandV).
// ---------------------------------------------------------------------------
__global__ __launch_bounds__(128) void attn_band_kernel(
    const float* __restrict__ qkv, const float* __restrict__ sumV,
    const int* __restrict__ epoch_p, float* __restrict__ attOut)
{
    __shared__ float pbuf[4][48];
    const int lane  = threadIdx.x & 31;
    const int wslot = threadIdx.x >> 5;
    const int gw = blockIdx.x * 4 + wslot;        // 0 .. B*H*N-1
    const int n = gw & (N_ - 1);
    const int h = (gw >> 11) & (H_ - 1);
    const int b = gw >> 15;

    const int w = (epoch_p[0] < 15) ? 16 : 20;    // faithful buggy-branch window
    const int count = 2 * w + 1;
    const int bandStart = n - w;
    const int lo = bandStart < 0 ? 0 : bandStart;
    const int hi = (n + w) > (N_ - 1) ? (N_ - 1) : (n + w);
    const int bandCnt = hi - lo + 1;

    const size_t rs = 3 * C_;
    const float* qrow = qkv + (size_t)(b * N_ + n) * rs + h * DH_;

    // 1) band logits (scale = Dh // H = 4)
    for (int idx = lane; idx < count; idx += 32) {
        int m = bandStart + idx;
        float logit = -1e30f;
        if (m >= 0 && m < N_) {
            const float* krow = qkv + (size_t)(b * N_ + m) * rs + C_ + h * DH_;
            float d = 0.f;
            #pragma unroll 8
            for (int e = 0; e < DH_; ++e) d += qrow[e] * krow[e];
            logit = 4.0f * d;
        }
        pbuf[wslot][idx] = logit;
    }
    __syncthreads();

    // 2) row max (includes the 1e-9 plateau of masked entries)
    float lm = -1e30f;
    for (int idx = lane; idx < count; idx += 32) lm = fmaxf(lm, pbuf[wslot][idx]);
    #pragma unroll
    for (int off = 16; off > 0; off >>= 1) lm = fmaxf(lm, __shfl_xor(lm, off, 32));
    const float Mx  = fmaxf(lm, 1e-9f);
    const float pnb = __expf(1e-9f - Mx);

    // 3) partition function
    float lz = 0.f;
    for (int idx = lane; idx < count; idx += 32) {
        float p = pbuf[wslot][idx];
        if (p > -1e29f) lz += __expf(p - Mx);
    }
    #pragma unroll
    for (int off = 16; off > 0; off >>= 1) lz += __shfl_xor(lz, off, 32);
    const float Z = lz + (float)(N_ - bandCnt) * pnb;
    __syncthreads();

    // 4) overwrite band logits with unnormalized probabilities
    for (int idx = lane; idx < count; idx += 32) {
        float p = pbuf[wslot][idx];
        pbuf[wslot][idx] = (p > -1e29f) ? __expf(p - Mx) : 0.f;
    }
    __syncthreads();

    // 5) out[e] = (sum_band p*v + pnb*(sumV - bandV)) / Z     (2 e's per lane)
    const float invZ = 1.0f / Z;
    #pragma unroll
    for (int t = 0; t < 2; ++t) {
        int e = lane + 32 * t;
        float acc = 0.f, bsum = 0.f;
        for (int idx = 0; idx < count; ++idx) {
            int m = bandStart + idx;
            if (m >= 0 && m < N_) {
                float ve = qkv[(size_t)(b * N_ + m) * rs + 2 * C_ + h * DH_ + e];
                acc  += pbuf[wslot][idx] * ve;
                bsum += ve;
            }
        }
        float sv = sumV[(b * H_ + h) * DH_ + e];
        attOut[(size_t)(b * N_ + n) * C_ + h * DH_ + e] = (acc + pnb * (sv - bsum)) * invZ;
    }
}

// ---------------------------------------------------------------------------
extern "C" void kernel_launch(void* const* d_in, const int* in_sizes, int n_in,
                              void* d_out, int out_size, void* d_ws, size_t ws_size,
                              hipStream_t stream)
{
    (void)in_sizes; (void)n_in; (void)out_size; (void)ws_size;

    const float* x      = (const float*)d_in[0];   // [B,N,C]
    const float* qkv_w  = (const float*)d_in[1];   // [3C,C]
    const float* proj_w = (const float*)d_in[2];   // [C,C]
    const float* proj_b = (const float*)d_in[3];   // [C]
    const int*   epoch  = (const int*)d_in[4];     // scalar (device-resident)

    float* qkv    = (float*)d_ws;                          // 4096*3072 f32 (50.3 MB)
    float* attOut = qkv + (size_t)B_ * N_ * 3 * C_;        // 4096*1024 f32 (16.8 MB)
    float* sumV   = attOut + (size_t)B_ * N_ * C_;         // 2048 f32
    float* out    = (float*)d_out;

    // qkv = x @ qkv_w^T : M=4096, N=3072, K=1024
    {
        dim3 grid((3 * C_) / 128, (B_ * N_) / 64);
        gemm_wmma_kernel<<<grid, 256, 0, stream>>>(x, qkv_w, nullptr, qkv,
                                                   B_ * N_, 3 * C_, C_);
    }
    // sumV over all keys' V rows (for the 1e-9-softmax rank-1 correction)
    sumv_kernel<<<(B_ * H_ * DH_ + 255) / 256, 256, 0, stream>>>(qkv, sumV);

    // band attention, one wave per (b,h,n)
    attn_band_kernel<<<(B_ * H_ * N_) / 4, 128, 0, stream>>>(qkv, sumV, epoch, attOut);

    // out = attOut @ proj_w^T + proj_b : M=4096, N=1024, K=1024
    {
        dim3 grid(C_ / 128, (B_ * N_) / 64);
        gemm_wmma_kernel<<<grid, 256, 0, stream>>>(attOut, proj_w, proj_b, out,
                                                   B_ * N_, C_, C_);
    }
}
